// QEqualizedConv0_19310172963284
// MI455X (gfx1250) — matrix operand order repaired and checked by reading, not compiled, not run on hardware
//
#include <hip/hip_runtime.h>
#include <hip/hip_bf16.h>

typedef __attribute__((ext_vector_type(16))) __bf16 v16bf;
typedef __attribute__((ext_vector_type(8)))  float  v8f;
typedef __attribute__((ext_vector_type(4)))  unsigned int v4u;
typedef __attribute__((ext_vector_type(8)))  int v8i;
typedef __attribute__((ext_vector_type(4)))  int v4i;
typedef unsigned short u16;

union FragBF {
  v16bf v;
  uint4 q[2];
};

__device__ __forceinline__ u16 f32_to_bf16(float f) {
  union { float f; unsigned u; } x;
  x.f = f;
  unsigned u = x.u;
  u += 0x7FFFu + ((u >> 16) & 1u);   // round-to-nearest-even
  return (u16)(u >> 16);
}

// ---- CDNA5 async copy: global -> LDS, 16B per lane (ASYNCcnt-tracked) ----
__device__ __forceinline__ void async_b128(void* lds_dst, const void* gsrc) {
  unsigned l = (unsigned)(unsigned long long)lds_dst;  // low 32b = LDS offset
  asm volatile("global_load_async_to_lds_b128 %0, %1, off"
               :: "v"(l), "v"(gsrc) : "memory");
}
__device__ __forceinline__ void wait_async0() {
  asm volatile("s_wait_asynccnt 0x0" ::: "memory");
}

// ---- CDNA5 Tensor Data Mover: one 2D tile (ROWSx32 bf16, row stride 1024)
// D# per CDNA5 ISA. One instruction per wave.
#define TDM_ROWS 256
__device__ __forceinline__ void tdm_load_a(const u16* gsrc, void* lds_dst) {
  unsigned long long ga = (unsigned long long)gsrc;
  unsigned lds = (unsigned)(unsigned long long)lds_dst;
  v4u g0;
  g0.x = 0x1u;                                            // count=1
  g0.y = lds;                                             // lds_addr
  g0.z = (unsigned)(ga & 0xFFFFFFFFu);                    // global_addr[31:0]
  g0.w = (unsigned)((ga >> 32) & 0x01FFFFFFu) | (2u << 30);  // addr[56:32], type=2
  v8i g1;
  g1[0] = 0x00010000;          // workgroup_mask=0, data_size=1 (2 bytes)
  g1[1] = (int)(1024u << 16);  // tensor_dim0 = 1024 (bits 79:48, low half)
  g1[2] = (int)(1024u << 16);  // tensor_dim0 hi=0; tensor_dim1 = 1024 (bits 111:80)
  g1[3] = (int)(32u << 16);    // tensor_dim1 hi=0; tile_dim0 = 32 (bits 127:112)
  g1[4] = TDM_ROWS;            // tile_dim1 = 256 rows; tile_dim2 = 0
  g1[5] = 1024;                // tensor_dim0_stride = 1024 (bits 207:160, low)
  g1[6] = 0;                   // stride hi = 0; tensor_dim1_stride lo = 0
  g1[7] = 0;
  v4i gz = {0, 0, 0, 0};
#if __clang_major__ >= 23
  v8i gz8 = {0, 0, 0, 0, 0, 0, 0, 0};
  __builtin_amdgcn_tensor_load_to_lds(g0, g1, gz, gz, gz8, 0);
#else
  __builtin_amdgcn_tensor_load_to_lds(g0, g1, gz, gz, 0);
#endif
}

#define DIM 1024
#define DIM2 (1024u * 1024u)
#define ABUF (256 * 32)   // halves per A LDS buffer (16 KB)
#define BBUF (128 * 32)   // halves per B LDS buffer (8 KB)

// ---------------------------------------------------------------------------
// Kernel 1: build U = G^{kron 5}; only transposed bf16 planes needed:
//   Ut_re = U_re^T (== A_re of U^dagger), Ut_im = U_im^T, Ut_imn = -U_im^T.
// ---------------------------------------------------------------------------
__global__ __launch_bounds__(256) void build_u_kernel(
    const float* __restrict__ w,
    u16* __restrict__ Ut_re, u16* __restrict__ Ut_im,
    u16* __restrict__ Ut_imn) {
  const unsigned idx = blockIdx.x * blockDim.x + threadIdx.x;
  const unsigned i = idx >> 10, j = idx & 1023u;

  const float WM = 0.63245553203367586f;  // sqrt(2/5)
  float h0 = w[0] * WM * 0.5f, h1 = w[1] * WM * 0.5f, h2 = w[2] * WM * 0.5f;
  float h3 = w[3] * WM * 0.5f, h4 = w[4] * WM * 0.5f, h5 = w[5] * WM * 0.5f;
  float c0 = cosf(h0), s0 = sinf(h0), c1 = cosf(h1), s1 = sinf(h1);
  float c2 = cosf(h2), s2 = sinf(h2), c3 = cosf(h3), s3 = sinf(h3);
  float c4 = cosf(h4), s4 = sinf(h4), c5 = cosf(h5), s5 = sinf(h5);

  float R0[2][2] = {{c0, -s0}, {s0, c0}};
  float R1[2][2] = {{c1, -s1}, {s1, c1}};
  float R3[2][2] = {{c3, -s3}, {s3, c3}};
  float R4[2][2] = {{c4, -s4}, {s4, c4}};

  float K1[4][4], K2[4][4];
#pragma unroll
  for (int a = 0; a < 2; ++a)
#pragma unroll
    for (int b = 0; b < 2; ++b)
#pragma unroll
      for (int c = 0; c < 2; ++c)
#pragma unroll
        for (int d = 0; d < 2; ++d) {
          K1[a * 2 + b][c * 2 + d] = R0[a][c] * R1[b][d];
          K2[a * 2 + b][c * 2 + d] = R3[a][c] * R4[b][d];
        }

  float M1r[4][4], M1i[4][4];
#pragma unroll
  for (int r = 0; r < 4; ++r)
#pragma unroll
    for (int c = 0; c < 4; ++c) {
      M1r[r][c] = c2 * K1[r][c];
      M1i[r][c] = -s2 * K1[3 - r][c];
    }
  float M2r[4][4], M2i[4][4];
#pragma unroll
  for (int r = 0; r < 4; ++r)
#pragma unroll
    for (int c = 0; c < 4; ++c) {
      float ar = 0.f, ai = 0.f;
#pragma unroll
      for (int k = 0; k < 4; ++k) {
        ar += K2[r][k] * M1r[k][c];
        ai += K2[r][k] * M1i[k][c];
      }
      M2r[r][c] = ar;
      M2i[r][c] = ai;
    }
  float Gr[4][4], Gi[4][4];
#pragma unroll
  for (int r = 0; r < 4; ++r)
#pragma unroll
    for (int c = 0; c < 4; ++c) {
      Gr[r][c] = c5 * M2r[r][c] + s5 * M2i[3 - r][c];
      Gi[r][c] = c5 * M2i[r][c] - s5 * M2r[3 - r][c];
    }

  float re = 1.f, im = 0.f;
#pragma unroll
  for (int p = 0; p < 5; ++p) {
    int sh = 8 - 2 * p;
    int ip = (i >> sh) & 3, jp = (j >> sh) & 3;
    float gr = Gr[ip][jp], gi = Gi[ip][jp];
    float nr = re * gr - im * gi;
    im = re * gi + im * gr;
    re = nr;
  }

  const size_t t = (size_t)j * DIM + i;  // transposed slot
  Ut_re[t]  = f32_to_bf16(re);
  Ut_im[t]  = f32_to_bf16(im);
  Ut_imn[t] = f32_to_bf16(-im);
}

// ---------------------------------------------------------------------------
// Kernel 2: f32 -> bf16 with per-batch transpose (padded-LDS 32x32 tiles).
// ---------------------------------------------------------------------------
__global__ __launch_bounds__(256) void cvt_t_kernel(const float* __restrict__ x,
                                                    u16* __restrict__ xT) {
  __shared__ u16 tile[32][33];
  const int b = blockIdx.z;
  const int r0 = blockIdx.y * 32, c0 = blockIdx.x * 32;
  const int tx = threadIdx.x & 31, ty = threadIdx.x >> 5;
  const float* xb = x + (size_t)b * DIM2;
  u16* xtb = xT + (size_t)b * DIM2;
#pragma unroll
  for (int l = 0; l < 4; ++l) {
    int r = ty + l * 8;
    tile[r][tx] = f32_to_bf16(xb[(size_t)(r0 + r) * DIM + c0 + tx]);
  }
  __syncthreads();
#pragma unroll
  for (int l = 0; l < 4; ++l) {
    int r = ty + l * 8;
    xtb[(size_t)(c0 + r) * DIM + r0 + tx] = tile[tx][r];
  }
}

// ---------------------------------------------------------------------------
// Staging one K-slab: A tile (256x32) via TDM from wave 0; B tile (128x32)
// via per-lane async b128 copies. Both row-major with K contiguous.
// ---------------------------------------------------------------------------
__device__ __forceinline__ void stage_tiles(const u16* __restrict__ A,
                                            const u16* __restrict__ Bt,
                                            int tileM, int tileN, int k0,
                                            u16* __restrict__ As,
                                            u16* __restrict__ Bs, int tid) {
  if ((tid >> 5) == 0) {  // wave 0: one TDM descriptor for the whole A tile
    tdm_load_a(A + (size_t)tileM * DIM + k0, As);
  }
  const int r = tid >> 1, sg = (tid & 1) * 16;
  const u16* gb = Bt + (size_t)(tileN + r) * DIM + k0 + sg;
  u16* lb = Bs + r * 32 + sg;
  async_b128(lb, gb);
  async_b128(lb + 8, gb + 8);
  __builtin_prefetch(gb + 32, 0, 3);  // next K-slab
}

// ---------------------------------------------------------------------------
// GEMM core: one 256x128 block tile, acc += A(256xK) @ Bt^T, both operands
// row-major with K contiguous (A[m][k], Bt[n][k]); double-buffered LDS;
// 8 waves as 4(M) x 2(N); wave tile 64x64 = 4x4 wmma tiles; K-step 32.
// ---------------------------------------------------------------------------
__device__ __forceinline__ void gemm_pass(const u16* __restrict__ A,
                                          const u16* __restrict__ Bt,
                                          int tileM, int tileN,
                                          u16* __restrict__ As,
                                          u16* __restrict__ Bs,
                                          v8f (&acc)[4][4]) {
  const int tid = threadIdx.x;
  const int lane = tid & 31, wid = tid >> 5;
  const int waveM = (wid & 3) * 64;   // 0,64,128,192
  const int waveN = (wid >> 2) * 64;  // 0,64

  int cur = 0;
  stage_tiles(A, Bt, tileM, tileN, 0, As, Bs, tid);
  wait_async0();
  __builtin_amdgcn_s_wait_tensorcnt(0);
  __syncthreads();

  for (int k0 = 0; k0 < DIM; k0 += 32) {
    // stage next slab into the other buffer (wraps harmlessly on last iter)
    const int kn = (k0 + 32) & (DIM - 1);
    stage_tiles(A, Bt, tileM, tileN, kn, As + (cur ^ 1) * ABUF,
                Bs + (cur ^ 1) * BBUF, tid);

    const u16* as = As + cur * ABUF;
    const u16* bs = Bs + cur * BBUF;

    FragBF afrag[4];
#pragma unroll
    for (int mi = 0; mi < 4; ++mi) {
      // A 16x32 bf16 frag: lane m = lane%16, h = lane/16;
      // VGPR0-3: K=h*8..h*8+7 ; VGPR4-7: K=16+h*8..16+h*8+7
      const u16* ap =
          as + (waveM + mi * 16 + (lane & 15)) * 32 + (lane >> 4) * 8;
      afrag[mi].q[0] = *(const uint4*)ap;
      afrag[mi].q[1] = *(const uint4*)(ap + 16);
    }
#pragma unroll
    for (int ni = 0; ni < 4; ++ni) {
      // B 32x16 bf16 frag: lane n = lane%16, h = lane/16;
      // VGPR0-3: K=h*8..h*8+7 ; VGPR4-7: K=16+h*8..16+h*8+7
      FragBF bfrag;
      const u16* bp =
          bs + (waveN + ni * 16 + (lane & 15)) * 32 + (lane >> 4) * 8;
      bfrag.q[0] = *(const uint4*)bp;
      bfrag.q[1] = *(const uint4*)(bp + 16);
#pragma unroll
      for (int mi = 0; mi < 4; ++mi) {
        acc[mi][ni] = __builtin_amdgcn_wmma_f32_16x16x32_bf16(
            false, afrag[mi].v, false, bfrag.v, (short)0, acc[mi][ni], false,
            false);
      }
    }
    wait_async0();
    __builtin_amdgcn_s_wait_tensorcnt(0);
    __syncthreads();
    cur ^= 1;
  }
}

// ---------------------------------------------------------------------------
// Stage 1: T_part = A_part @ x_b ; A_re = Ut_re, A_im = Ut_imn; B = xT_b.
// grid = (8, 4, 16); batch = z&7, part = z>>3
// ---------------------------------------------------------------------------
__global__ __launch_bounds__(256) void stage1_kernel(
    const u16* __restrict__ Ut_re, const u16* __restrict__ Ut_imn,
    const u16* __restrict__ xT, u16* __restrict__ T) {
  __shared__ u16 As[2 * ABUF];
  __shared__ u16 Bs[2 * BBUF];
  const int z = blockIdx.z, batch = z & 7, part = z >> 3;
  const u16* A  = part ? Ut_imn : Ut_re;
  const u16* Bt = xT + (size_t)batch * DIM2;
  u16* C = T + ((size_t)batch * 2 + part) * DIM2;
  const int tileM = blockIdx.y * 256, tileN = blockIdx.x * 128;

  v8f acc[4][4];
#pragma unroll
  for (int mi = 0; mi < 4; ++mi)
#pragma unroll
    for (int ni = 0; ni < 4; ++ni) acc[mi][ni] = (v8f)(0.f);

  gemm_pass(A, Bt, tileM, tileN, As, Bs, acc);

  const int lane = threadIdx.x & 31, wid = threadIdx.x >> 5;
  const int wm = tileM + (wid & 3) * 64, wn = tileN + (wid >> 2) * 64;
#pragma unroll
  for (int mi = 0; mi < 4; ++mi)
#pragma unroll
    for (int ni = 0; ni < 4; ++ni)
#pragma unroll
      for (int r = 0; r < 8; ++r) {
        // C/D layout: VGPR r -> M = r (lanes 0-15) or r+8 (lanes 16-31)
        int row = wm + mi * 16 + r + ((lane >> 4) << 3);
        int col = wn + ni * 16 + (lane & 15);
        C[(size_t)row * DIM + col] = f32_to_bf16(acc[mi][ni][r]);
      }
}

// ---------------------------------------------------------------------------
// Stage 2: out_re = T_re@U_re + T_im@(-U_im);  out_im = T_re@U_im + T_im@U_re
// B operands pre-transposed (Ut_*). grid = (8, 4, 16)
// ---------------------------------------------------------------------------
__global__ __launch_bounds__(256) void stage2_kernel(
    const u16* __restrict__ T, const u16* __restrict__ Ut_re,
    const u16* __restrict__ Ut_im, const u16* __restrict__ Ut_imn,
    float* __restrict__ out) {
  __shared__ u16 As[2 * ABUF];
  __shared__ u16 Bs[2 * BBUF];
  const int z = blockIdx.z, batch = z & 7, part = z >> 3;
  const u16* Tre = T + ((size_t)batch * 2 + 0) * DIM2;
  const u16* Tim = T + ((size_t)batch * 2 + 1) * DIM2;
  const u16* B0 = part ? Ut_im : Ut_re;
  const u16* B1 = part ? Ut_re : Ut_imn;
  const int tileM = blockIdx.y * 256, tileN = blockIdx.x * 128;

  v8f acc[4][4];
#pragma unroll
  for (int mi = 0; mi < 4; ++mi)
#pragma unroll
    for (int ni = 0; ni < 4; ++ni) acc[mi][ni] = (v8f)(0.f);

  gemm_pass(Tre, B0, tileM, tileN, As, Bs, acc);
  gemm_pass(Tim, B1, tileM, tileN, As, Bs, acc);

  float* C = out + ((size_t)part * 8 + batch) * DIM2;
  const int lane = threadIdx.x & 31, wid = threadIdx.x >> 5;
  const int wm = tileM + (wid & 3) * 64, wn = tileN + (wid >> 2) * 64;
#pragma unroll
  for (int mi = 0; mi < 4; ++mi)
#pragma unroll
    for (int ni = 0; ni < 4; ++ni)
#pragma unroll
      for (int r = 0; r < 8; ++r) {
        int row = wm + mi * 16 + r + ((lane >> 4) << 3);
        int col = wn + ni * 16 + (lane & 15);
        C[(size_t)row * DIM + col] = acc[mi][ni][r];
      }
}

// ---------------------------------------------------------------------------
extern "C" void kernel_launch(void* const* d_in, const int* in_sizes, int n_in,
                              void* d_out, int out_size, void* d_ws,
                              size_t ws_size, hipStream_t stream) {
  const float* x = (const float*)d_in[0];   // (8, 1024, 1024) f32
  const float* w = (const float*)d_in[1];   // (6,) f32

  // workspace (bf16 halves; 1 unit = 1M halves = 2 MB); total 54 MB
  u16* ws = (u16*)d_ws;
  const size_t U = (size_t)DIM2;
  u16* Ut_re  = ws + 0 * U;
  u16* Ut_im  = ws + 1 * U;
  u16* Ut_imn = ws + 2 * U;
  u16* xT     = ws + 3 * U;   // 8 * U
  u16* T      = ws + 11 * U;  // 16 * U

  build_u_kernel<<<DIM2 / 256, 256, 0, stream>>>(w, Ut_re, Ut_im, Ut_imn);
  cvt_t_kernel<<<dim3(32, 32, 8), 256, 0, stream>>>(x, xT);

  dim3 grid(8, 4, 16);
  stage1_kernel<<<grid, 256, 0, stream>>>(Ut_re, Ut_imn, xT, T);
  stage2_kernel<<<grid, 256, 0, stream>>>(T, Ut_re, Ut_im, Ut_imn,
                                          (float*)d_out);
}
